// GNN_23965917511968
// MI455X (gfx1250) — matrix operand order repaired and checked by reading, not compiled, not run on hardware
//
#include <hip/hip_runtime.h>
#include <hip/hip_bf16.h>

// ---------------------------------------------------------------------------
// Types for WMMA (CDNA5 gfx1250, wave32)
// ---------------------------------------------------------------------------
typedef __bf16          v16bf  __attribute__((ext_vector_type(16)));
typedef unsigned short  v16us  __attribute__((ext_vector_type(16)));
typedef unsigned short  ushort8 __attribute__((ext_vector_type(8)));
typedef float           v8f    __attribute__((ext_vector_type(8)));

#define D128 128
#define SCALE_D 0.08838834764831845f  // 1/sqrt(128)

// ---------------------------------------------------------------------------
// bf16 split helpers (round-to-nearest-even, bit tricks)
// ---------------------------------------------------------------------------
__device__ __forceinline__ unsigned short f2bf(float x) {
    unsigned u = __float_as_uint(x);
    unsigned r = u + 0x7fffu + ((u >> 16) & 1u);
    return (unsigned short)(r >> 16);
}
__device__ __forceinline__ float bf2f(unsigned short h) {
    return __uint_as_float(((unsigned)h) << 16);
}

// monotonic float<->uint encoding for atomicMax on floats (0 == -inf-ish)
__device__ __forceinline__ unsigned encf(float f) {
    unsigned u = __float_as_uint(f);
    return (u & 0x80000000u) ? ~u : (u | 0x80000000u);
}
__device__ __forceinline__ float decf(unsigned u) {
    u = (u & 0x80000000u) ? (u & 0x7fffffffu) : ~u;
    return __uint_as_float(u);
}

// ---------------------------------------------------------------------------
// Split a f32 tensor into bf16 hi/lo planes (one pass; removes all conversion
// VALU from the GEMM hot loops).
// ---------------------------------------------------------------------------
__global__ void __launch_bounds__(256) k_split_x(const float* __restrict__ X,
                                                 unsigned short* __restrict__ Xh,
                                                 unsigned short* __restrict__ Xl,
                                                 long long total) {
    long long i = (long long)blockIdx.x * blockDim.x + threadIdx.x;
    if (i >= total) return;
    float v = X[i];
    unsigned short h = f2bf(v);
    Xh[i] = h;
    Xl[i] = f2bf(v - bf2f(h));
}

// ---------------------------------------------------------------------------
// Core: one wave computes a 16(row) x 16(col) f32 tile of  X(16x128) @ W(128xC)
// using bf16x3 split precision (Ah*Bh + Al*Bh + Ah*Bl) on v_wmma_f32_16x16x32_bf16.
//   Xh/Xl   : bf16 hi/lo planes of X, row-major, ld = 128
//   WtH/WtL : bf16 hi/lo of W, TRANSPOSED: Wt[outcol][k], ld = 128
// A layout (16x32 bf16): lane m<16 holds row m, K elems {k0..k0+7, k0+16..k0+23};
//                        lane m+16 holds row m, K elems {k0+8..15, k0+24..31}.
// B layout (32x16 bf16): lane n<16 holds col n, K elems {k0..k0+7, k0+16..k0+23};
//                        lane n+16 holds col n, K elems {k0+8..15, k0+24..31}.
// ---------------------------------------------------------------------------
__device__ __forceinline__ v8f gemm16_tile(const unsigned short* __restrict__ Xh,
                                           const unsigned short* __restrict__ Xl,
                                           long long row0,
                                           const unsigned short* __restrict__ WtH,
                                           const unsigned short* __restrict__ WtL,
                                           int col0) {
    const int lane = threadIdx.x & 31;
    const int m    = lane & 15;
    const int hi16 = lane >> 4;
    v8f acc = {};
    const unsigned short* xh = Xh + (row0 + m) * D128;
    const unsigned short* xl = Xl + (row0 + m) * D128;
    const int col = col0 + m;
#pragma unroll
    for (int k0 = 0; k0 < 128; k0 += 32) {
        const int base = k0 + hi16 * 8;
        // ---- A operand: two 16B loads each for hi and lo planes
        ushort8 ah0 = *(const ushort8*)(xh + base);
        ushort8 ah1 = *(const ushort8*)(xh + base + 16);
        ushort8 al0 = *(const ushort8*)(xl + base);
        ushort8 al1 = *(const ushort8*)(xl + base + 16);
        v16us ah = __builtin_shufflevector(ah0, ah1, 0, 1, 2, 3, 4, 5, 6, 7,
                                           8, 9, 10, 11, 12, 13, 14, 15);
        v16us al = __builtin_shufflevector(al0, al1, 0, 1, 2, 3, 4, 5, 6, 7,
                                           8, 9, 10, 11, 12, 13, 14, 15);
        // ---- B operand: two 16B bf16 loads each from pre-transposed weights
        const unsigned short* wb = WtH + (size_t)col * 128 + base;
        const unsigned short* wl = WtL + (size_t)col * 128 + base;
        v16us bh = __builtin_shufflevector(*(const ushort8*)wb, *(const ushort8*)(wb + 16),
                                           0, 1, 2, 3, 4, 5, 6, 7, 8, 9, 10, 11, 12, 13, 14, 15);
        v16us bl = __builtin_shufflevector(*(const ushort8*)wl, *(const ushort8*)(wl + 16),
                                           0, 1, 2, 3, 4, 5, 6, 7, 8, 9, 10, 11, 12, 13, 14, 15);

        v16bf Ah = __builtin_bit_cast(v16bf, ah);
        v16bf Al = __builtin_bit_cast(v16bf, al);
        v16bf Bh = __builtin_bit_cast(v16bf, bh);
        v16bf Bl = __builtin_bit_cast(v16bf, bl);

        acc = __builtin_amdgcn_wmma_f32_16x16x32_bf16(false, Ah, false, Bh, (short)0, acc, false, false);
        acc = __builtin_amdgcn_wmma_f32_16x16x32_bf16(false, Al, false, Bh, (short)0, acc, false, false);
        acc = __builtin_amdgcn_wmma_f32_16x16x32_bf16(false, Ah, false, Bl, (short)0, acc, false, false);
    }
    return acc;
}

// ---------------------------------------------------------------------------
// GEMM: KV = X @ [Wk|Wv] + [bk|bv]   (output ld = 256; K cols 0..127, V 128..255)
// grid (N/16, 2), block 256 (8 waves). Wave w -> cols blockIdx.y*128 + w*16.
// ---------------------------------------------------------------------------
__global__ void __launch_bounds__(256) k_gemm_kv(const unsigned short* __restrict__ Xh,
                                                 const unsigned short* __restrict__ Xl,
                                                 const unsigned short* __restrict__ WtH,
                                                 const unsigned short* __restrict__ WtL,
                                                 const float* __restrict__ bias,
                                                 float* __restrict__ Y) {
    const int wave = threadIdx.x >> 5;
    const long long row0 = (long long)blockIdx.x * 16;
    const int col0 = blockIdx.y * 128 + wave * 16;
    v8f acc = gemm16_tile(Xh, Xl, row0, WtH, WtL, col0);
    const int lane = threadIdx.x & 31;
    const int m = lane & 15, hi16 = lane >> 4;
    const int col = col0 + m;
    const float b = bias[col];
#pragma unroll
    for (int j = 0; j < 8; ++j) {
        long long row = row0 + j + hi16 * 8;
        Y[row * 256 + col] = acc[j] + b;
    }
}

// ---------------------------------------------------------------------------
// GEMM Q + fused dots: Q = X @ Wq + bq kept in LDS only; outputs
//   s_self[n] = (Q[n]·K[n]) * scale     (K read from KV buffer, ld 256)
//   s_edge[n] = (Q[n]·r)    * scale     (r = rel @ ksum, 128 floats)
// grid N/16, block 256 (8 waves, wave w -> cols w*16).
// ---------------------------------------------------------------------------
__global__ void __launch_bounds__(256) k_gemm_q_dots(const unsigned short* __restrict__ Xh,
                                                     const unsigned short* __restrict__ Xl,
                                                     const unsigned short* __restrict__ WtH,
                                                     const unsigned short* __restrict__ WtL,
                                                     const float* __restrict__ bias,
                                                     const float* __restrict__ KV,
                                                     const float* __restrict__ r,
                                                     float* __restrict__ s_self,
                                                     float* __restrict__ s_edge) {
    __shared__ float q[16][130];
    const int wave = threadIdx.x >> 5;
    const int lane = threadIdx.x & 31;
    const long long row0 = (long long)blockIdx.x * 16;
    const int col0 = wave * 16;
    v8f acc = gemm16_tile(Xh, Xl, row0, WtH, WtL, col0);
    const int m = lane & 15, hi16 = lane >> 4;
    const int col = col0 + m;
    const float b = bias[col];
#pragma unroll
    for (int j = 0; j < 8; ++j)
        q[j + hi16 * 8][col] = acc[j] + b;
    __syncthreads();

    // dot phase: thread t -> (row = t/16, 8-col chunk = t%16)
    const int row = threadIdx.x >> 4;
    const int chunk = threadIdx.x & 15;
    const float* Krow = KV + (row0 + row) * 256;  // K part = cols 0..127
    float p1 = 0.f, p2 = 0.f;
#pragma unroll
    for (int j = 0; j < 8; ++j) {
        int c = chunk * 8 + j;
        float qv = q[row][c];
        p1 += qv * Krow[c];
        p2 += qv * r[c];
    }
#pragma unroll
    for (int off = 8; off; off >>= 1) {
        p1 += __shfl_xor(p1, off, 32);
        p2 += __shfl_xor(p2, off, 32);
    }
    if (chunk == 0) {
        s_self[row0 + row] = p1 * SCALE_D;
        s_edge[row0 + row] = p2 * SCALE_D;
    }
}

// ---------------------------------------------------------------------------
// Final residual GEMM:  Y = Y + X @ W + b   (Y: N x 128; Xh/Xl are the bf16
// split of Y taken before this kernel, so the only in-place access is each
// lane's own read-then-write of its output element).
// ---------------------------------------------------------------------------
__global__ void __launch_bounds__(256) k_gemm_self(float* __restrict__ Y,
                                                   const unsigned short* __restrict__ Xh,
                                                   const unsigned short* __restrict__ Xl,
                                                   const unsigned short* __restrict__ WtH,
                                                   const unsigned short* __restrict__ WtL,
                                                   const float* __restrict__ bias) {
    const int wave = threadIdx.x >> 5;
    const int lane = threadIdx.x & 31;
    const long long row0 = (long long)blockIdx.x * 16;
    const int col0 = wave * 16;
    v8f acc = gemm16_tile(Xh, Xl, row0, WtH, WtL, col0);
    const int m = lane & 15, hi16 = lane >> 4;
    const int col = col0 + m;
    const float b = bias[col];
#pragma unroll
    for (int j = 0; j < 8; ++j) {
        long long idx = (row0 + j + hi16 * 8) * D128 + col;
        Y[idx] = Y[idx] + acc[j] + b;
    }
}

// ---------------------------------------------------------------------------
// Weight prep: transpose + bf16 hi/lo split, once per call.
//   qkv_w [2][2][3][128][128] (in,out) -> WtQ[lt][128out][128in], WtKV[lt][256out][128in]
//   self_w [2][128][128]               -> WtS[t][128out][128in]
// ---------------------------------------------------------------------------
__global__ void k_prep_weights(const float* __restrict__ qkv_w, const float* __restrict__ qkv_b,
                               const float* __restrict__ self_w, const float* __restrict__ self_b,
                               unsigned short* WtQ_h, unsigned short* WtQ_l,
                               unsigned short* WtKV_h, unsigned short* WtKV_l,
                               unsigned short* WtS_h, unsigned short* WtS_l,
                               float* biasQ, float* biasKV, float* biasS) {
    const int tid = blockIdx.x * blockDim.x + threadIdx.x;
    const int stride = gridDim.x * blockDim.x;
    const int QKV = 2 * 2 * 3 * 128 * 128;
    for (int idx = tid; idx < QKV; idx += stride) {
        int out = idx & 127;
        int in  = (idx >> 7) & 127;
        int j   = (idx / (128 * 128)) % 3;
        int lt  = idx / (3 * 128 * 128);  // l*2 + t
        float v = qkv_w[idx];
        unsigned short h = f2bf(v);
        unsigned short lo = f2bf(v - bf2f(h));
        if (j == 0) {
            int d = (lt * 128 + out) * 128 + in;
            WtQ_h[d] = h; WtQ_l[d] = lo;
        } else {
            int d = (lt * 256 + (j - 1) * 128 + out) * 128 + in;
            WtKV_h[d] = h; WtKV_l[d] = lo;
        }
    }
    const int SW = 2 * 128 * 128;
    for (int idx = tid; idx < SW; idx += stride) {
        int out = idx & 127;
        int in  = (idx >> 7) & 127;
        int t   = idx >> 14;
        float v = self_w[idx];
        unsigned short h = f2bf(v);
        unsigned short lo = f2bf(v - bf2f(h));
        int d = (t * 128 + out) * 128 + in;
        WtS_h[d] = h; WtS_l[d] = lo;
    }
    const int QB = 2 * 2 * 3 * 128;
    for (int idx = tid; idx < QB; idx += stride) {
        int o = idx & 127;
        int j = (idx >> 7) % 3;
        int lt = idx / (3 * 128);
        float v = qkv_b[idx];
        if (j == 0) biasQ[lt * 128 + o] = v;
        else        biasKV[lt * 256 + (j - 1) * 128 + o] = v;
    }
    for (int idx = tid; idx < 2 * 128; idx += stride) biasS[idx] = self_b[idx];
}

// ---------------------------------------------------------------------------
// Degree count, ksum, matvec, softmax reductions, edge passes
// ---------------------------------------------------------------------------
__global__ void k_count_deg(const int* __restrict__ dst, int* __restrict__ deg, int E) {
    int e = blockIdx.x * blockDim.x + threadIdx.x;
    if (e < E) atomicAdd(&deg[dst[e]], 1);
}

// ksum[c] = sum_n deg[n] * K[n][c]   (K = KV cols 0..127, ld 256)
__global__ void __launch_bounds__(256) k_ksum(const float* __restrict__ KV,
                                              const int* __restrict__ deg,
                                              float* __restrict__ ksum, int N) {
    const int c = threadIdx.x & 127;
    const int half = threadIdx.x >> 7;
    float acc = 0.f;
    for (int n = blockIdx.x * 2 + half; n < N; n += gridDim.x * 2)
        acc += (float)deg[n] * KV[(long long)n * 256 + c];
    __shared__ float sh[128];
    if (half) sh[c] = acc;
    __syncthreads();
    if (!half) atomicAdd(&ksum[c], acc + sh[c]);
}

// y = W @ x  for 128x128 W (row-major)
__global__ void k_matvec128(const float* __restrict__ W, const float* __restrict__ x,
                            float* __restrict__ y) {
    int m = threadIdx.x;
    float acc = 0.f;
#pragma unroll 4
    for (int o = 0; o < 128; ++o) acc += W[m * 128 + o] * x[o];
    y[m] = acc;
}

__global__ void __launch_bounds__(256) k_reduce_max(const float* __restrict__ s, int N,
                                                    unsigned* __restrict__ Mout) {
    float mx = -3.0e38f;
    for (int i = blockIdx.x * blockDim.x + threadIdx.x; i < N; i += gridDim.x * blockDim.x)
        mx = fmaxf(mx, s[i]);
#pragma unroll
    for (int off = 16; off; off >>= 1) mx = fmaxf(mx, __shfl_xor(mx, off, 32));
    __shared__ float sh[8];
    if ((threadIdx.x & 31) == 0) sh[threadIdx.x >> 5] = mx;
    __syncthreads();
    if (threadIdx.x == 0) {
        float mm = sh[0];
#pragma unroll
        for (int w = 1; w < 8; ++w) mm = fmaxf(mm, sh[w]);
        atomicMax(Mout, encf(mm));
    }
}

__global__ void __launch_bounds__(256) k_reduce_sumexp(const float* __restrict__ s, int N,
                                                       const unsigned* __restrict__ Mp,
                                                       float* __restrict__ Sout) {
    const float M = decf(*Mp);
    float sum = 0.f;
    for (int i = blockIdx.x * blockDim.x + threadIdx.x; i < N; i += gridDim.x * blockDim.x)
        sum += expf(s[i] - M);
#pragma unroll
    for (int off = 16; off; off >>= 1) sum += __shfl_xor(sum, off, 32);
    __shared__ float sh[8];
    if ((threadIdx.x & 31) == 0) sh[threadIdx.x >> 5] = sum;
    __syncthreads();
    if (threadIdx.x == 0) {
        float ss = sh[0];
#pragma unroll
        for (int w = 1; w < 8; ++w) ss += sh[w];
        atomicAdd(Sout, ss);
    }
}

__global__ void k_edge_max(const int* __restrict__ src, const int* __restrict__ dst,
                           const float* __restrict__ s, unsigned* __restrict__ m, int E) {
    int e = blockIdx.x * blockDim.x + threadIdx.x;
    if (e < E) atomicMax(&m[dst[e]], encf(s[src[e]]));
}

__global__ void k_edge_z(const int* __restrict__ src, const int* __restrict__ dst,
                         const float* __restrict__ s, const unsigned* __restrict__ m,
                         float* __restrict__ z, int E) {
    int e = blockIdx.x * blockDim.x + threadIdx.x;
    if (e < E) atomicAdd(&z[dst[e]], expf(s[src[e]] - decf(m[dst[e]])));
}

// out[dst] += (exp(s[src]-m[dst]) / (z[dst]*max(deg,1))) * V[src]
// 128 threads per edge; V = KV cols 128..255.
__global__ void __launch_bounds__(256) k_scatter(const int* __restrict__ src,
                                                 const int* __restrict__ dst,
                                                 const float* __restrict__ s,
                                                 const unsigned* __restrict__ m,
                                                 const float* __restrict__ z,
                                                 const int* __restrict__ deg,
                                                 const float* __restrict__ KVsrc,
                                                 float* __restrict__ out, int E) {
    long long idx = (long long)blockIdx.x * blockDim.x + threadIdx.x;
    int e = (int)(idx >> 7);
    int c = (int)(idx & 127);
    if (e >= E) return;
    int sn = src[e], dn = dst[e];
    float coeff = expf(s[sn] - decf(m[dn])) / (z[dn] * fmaxf((float)deg[dn], 1.0f));
    atomicAdd(&out[(long long)dn * 128 + c], coeff * KVsrc[(long long)sn * 256 + 128 + c]);
}

// out[n][c] += softmax_self(n) * V[n][c]
__global__ void __launch_bounds__(256) k_combine_self(float* __restrict__ out,
                                                      const float* __restrict__ KV,
                                                      const float* __restrict__ s_self,
                                                      const unsigned* __restrict__ Mp,
                                                      const float* __restrict__ Sp, int N) {
    long long idx = (long long)blockIdx.x * blockDim.x + threadIdx.x;
    if (idx >= (long long)N * 128) return;
    int n = (int)(idx >> 7);
    float a = expf(s_self[n] - decf(*Mp)) / (*Sp);
    out[idx] += a * KV[(long long)n * 256 + 128 + (idx & 127)];
}

// ---------------------------------------------------------------------------
// Host launch
// ---------------------------------------------------------------------------
extern "C" void kernel_launch(void* const* d_in, const int* in_sizes, int n_in,
                              void* d_out, int out_size, void* d_ws, size_t ws_size,
                              hipStream_t stream) {
    const float* x_user = (const float*)d_in[0];
    const float* x_item = (const float*)d_in[1];
    const int*   ei_u2i = (const int*)d_in[2];
    const int*   ei_i2u = (const int*)d_in[3];
    const float* qkv_w  = (const float*)d_in[4];
    const float* qkv_b  = (const float*)d_in[5];
    const float* rel_w  = (const float*)d_in[6];
    const float* self_w = (const float*)d_in[7];
    const float* self_b = (const float*)d_in[8];

    const int N = in_sizes[0] / 128;   // 100000 (multiple of 16)
    const int E = in_sizes[2] / 2;     // 600000

    const int* src_u2i = ei_u2i;       const int* dst_u2i = ei_u2i + E;
    const int* src_i2u = ei_i2u;       const int* dst_i2u = ei_i2u + E;

    float* outU = (float*)d_out;
    float* outI = outU + (size_t)N * 128;

    // ---- workspace carve-up
    char* w = (char*)d_ws;
    size_t off = 0;
    auto take = [&](size_t bytes) -> void* {
        void* p = w + off;
        off = (off + bytes + 255) & ~(size_t)255;
        return p;
    };
    float*    KV_u   = (float*)take((size_t)N * 256 * 4);
    float*    KV_i   = (float*)take((size_t)N * 256 * 4);
    float*    A_u    = (float*)take((size_t)N * 128 * 4);  // layer1 output / layer2 input
    float*    A_i    = (float*)take((size_t)N * 128 * 4);
    unsigned short* Xh_u = (unsigned short*)take((size_t)N * 128 * 2);  // bf16 hi/lo planes
    unsigned short* Xl_u = (unsigned short*)take((size_t)N * 128 * 2);
    unsigned short* Xh_i = (unsigned short*)take((size_t)N * 128 * 2);
    unsigned short* Xl_i = (unsigned short*)take((size_t)N * 128 * 2);
    float*    s_self_u = (float*)take((size_t)N * 4);
    float*    s_self_i = (float*)take((size_t)N * 4);
    float*    s_edge_u = (float*)take((size_t)N * 4);
    float*    s_edge_i = (float*)take((size_t)N * 4);
    int*      deg_u  = (int*)take((size_t)N * 4 * 2);      // deg_u then deg_i contiguous
    int*      deg_i  = deg_u + N;
    unsigned* m_u    = (unsigned*)take((size_t)N * 4 * 4); // m_u,z_u,m_i,z_i contiguous
    float*    z_u    = (float*)(m_u + N);
    unsigned* m_i    = (unsigned*)(m_u + 2 * (size_t)N);
    float*    z_i    = (float*)(m_u + 3 * (size_t)N);
    float*    stats  = (float*)take(4096);
    float*    ksum_u = stats + 0;
    float*    ksum_i = stats + 128;
    float*    r_items = stats + 256;
    float*    r_users = stats + 384;
    unsigned* Mu = (unsigned*)(stats + 512);
    float*    Su = stats + 513;
    unsigned* Mi = (unsigned*)(stats + 514);
    float*    Si = stats + 515;
    unsigned short* WtQ_h  = (unsigned short*)take(2 * 2 * 128 * 128 * 2);
    unsigned short* WtQ_l  = (unsigned short*)take(2 * 2 * 128 * 128 * 2);
    unsigned short* WtKV_h = (unsigned short*)take(2 * 2 * 256 * 128 * 2);
    unsigned short* WtKV_l = (unsigned short*)take(2 * 2 * 256 * 128 * 2);
    unsigned short* WtS_h  = (unsigned short*)take(2 * 128 * 128 * 2);
    unsigned short* WtS_l  = (unsigned short*)take(2 * 128 * 128 * 2);
    float* biasQ  = (float*)take(2 * 2 * 128 * 4);
    float* biasKV = (float*)take(2 * 2 * 256 * 4);
    float* biasS  = (float*)take(2 * 128 * 4);
    (void)ws_size; (void)n_in; (void)out_size;

    const long long nd = (long long)N * 128;
    const int gN16   = N / 16;
    const int gEdge  = (E + 255) / 256;
    const int gScat  = (int)(((long long)E * 128 + 255) / 256);
    const int gElem  = (int)((nd + 255) / 256);

    // ---- weight prep + degrees (per call; deterministic)
    k_prep_weights<<<256, 256, 0, stream>>>(qkv_w, qkv_b, self_w, self_b,
                                            WtQ_h, WtQ_l, WtKV_h, WtKV_l, WtS_h, WtS_l,
                                            biasQ, biasKV, biasS);
    hipMemsetAsync(deg_u, 0, (size_t)N * 4 * 2, stream);
    k_count_deg<<<gEdge, 256, 0, stream>>>(dst_i2u, deg_u, E);
    k_count_deg<<<gEdge, 256, 0, stream>>>(dst_u2i, deg_i, E);

    for (int l = 0; l < 2; ++l) {
        const float* Xu = l ? A_u : x_user;
        const float* Xi = l ? A_i : x_item;
        float* Ou = l ? outU : A_u;
        float* Oi = l ? outI : A_i;
        const int lt_u = l * 2 + 0, lt_i = l * 2 + 1;

        // zero per-layer accumulators (all encodings are zero-init friendly)
        hipMemsetAsync(m_u, 0, (size_t)N * 4 * 4, stream);
        hipMemsetAsync(stats, 0, 4096, stream);
        if (l == 0) hipMemsetAsync(A_u, 0, (size_t)N * 128 * 4 * 2, stream);
        else        hipMemsetAsync(d_out, 0, (size_t)N * 128 * 4 * 2, stream);

        // bf16 hi/lo split of current layer inputs (once per element)
        k_split_x<<<gElem, 256, 0, stream>>>(Xu, Xh_u, Xl_u, nd);
        k_split_x<<<gElem, 256, 0, stream>>>(Xi, Xh_i, Xl_i, nd);

        // K,V projections (WMMA bf16x3)
        k_gemm_kv<<<dim3(gN16, 2), 256, 0, stream>>>(
            Xh_u, Xl_u, WtKV_h + (size_t)lt_u * 256 * 128, WtKV_l + (size_t)lt_u * 256 * 128,
            biasKV + lt_u * 256, KV_u);
        k_gemm_kv<<<dim3(gN16, 2), 256, 0, stream>>>(
            Xh_i, Xl_i, WtKV_h + (size_t)lt_i * 256 * 128, WtKV_l + (size_t)lt_i * 256 * 128,
            biasKV + lt_i * 256, KV_i);

        // ksum = deg-weighted K reduction; r = rel @ ksum
        k_ksum<<<512, 256, 0, stream>>>(KV_u, deg_u, ksum_u, N);
        k_ksum<<<512, 256, 0, stream>>>(KV_i, deg_i, ksum_i, N);
        k_matvec128<<<1, 128, 0, stream>>>(rel_w + ((size_t)l * 2 + 1) * 128 * 128, ksum_u, r_items);
        k_matvec128<<<1, 128, 0, stream>>>(rel_w + ((size_t)l * 2 + 0) * 128 * 128, ksum_i, r_users);

        // Q projection fused with both score dots (Q never leaves LDS)
        k_gemm_q_dots<<<gN16, 256, 0, stream>>>(
            Xh_u, Xl_u, WtQ_h + (size_t)lt_u * 128 * 128, WtQ_l + (size_t)lt_u * 128 * 128,
            biasQ + lt_u * 128, KV_u, r_users, s_self_u, s_edge_u);
        k_gemm_q_dots<<<gN16, 256, 0, stream>>>(
            Xh_i, Xl_i, WtQ_h + (size_t)lt_i * 128 * 128, WtQ_l + (size_t)lt_i * 128 * 128,
            biasQ + lt_i * 128, KV_i, r_items, s_self_i, s_edge_i);

        // self softmax stats (global over nodes)
        k_reduce_max<<<256, 256, 0, stream>>>(s_self_u, N, Mu);
        k_reduce_max<<<256, 256, 0, stream>>>(s_self_i, N, Mi);
        k_reduce_sumexp<<<256, 256, 0, stream>>>(s_self_u, N, Mu, Su);
        k_reduce_sumexp<<<256, 256, 0, stream>>>(s_self_i, N, Mi, Si);

        // segment softmax stats per dst
        k_edge_max<<<gEdge, 256, 0, stream>>>(src_i2u, dst_i2u, s_edge_i, m_u, E);
        k_edge_max<<<gEdge, 256, 0, stream>>>(src_u2i, dst_u2i, s_edge_u, m_i, E);
        k_edge_z<<<gEdge, 256, 0, stream>>>(src_i2u, dst_i2u, s_edge_i, m_u, z_u, E);
        k_edge_z<<<gEdge, 256, 0, stream>>>(src_u2i, dst_u2i, s_edge_u, m_i, z_i, E);

        // edge message scatter (atomic f32)
        k_scatter<<<gScat, 256, 0, stream>>>(src_i2u, dst_i2u, s_edge_i, m_u, z_u, deg_u, KV_i, Ou, E);
        k_scatter<<<gScat, 256, 0, stream>>>(src_u2i, dst_u2i, s_edge_u, m_i, z_i, deg_i, KV_u, Oi, E);

        // add self-attention message
        k_combine_self<<<gElem, 256, 0, stream>>>(Ou, KV_u, s_self_u, Mu, Su, N);
        k_combine_self<<<gElem, 256, 0, stream>>>(Oi, KV_i, s_self_i, Mi, Si, N);
    }

    // final residual projections (A operand from fresh bf16 split of d_out)
    k_split_x<<<gElem, 256, 0, stream>>>(outU, Xh_u, Xl_u, nd);
    k_split_x<<<gElem, 256, 0, stream>>>(outI, Xh_i, Xl_i, nd);
    k_gemm_self<<<gN16, 256, 0, stream>>>(outU, Xh_u, Xl_u, WtS_h, WtS_l, biasS);
    k_gemm_self<<<gN16, 256, 0, stream>>>(outI, Xh_i, Xl_i,
                                          WtS_h + (size_t)128 * 128, WtS_l + (size_t)128 * 128,
                                          biasS + 128);
}
